// FuelBurnPredictor_16020228014425
// MI455X (gfx1250) — matrix operand order repaired and checked by reading, not compiled, not run on hardware
//
#include <hip/hip_runtime.h>
#include <hip/hip_bf16.h>

#define T_TOTAL 32768
#define NSEQ    16
#define INDIM   16
#define HIDN    512
#define NH      8
#define DKK     48
#define DVV     96
#define KD      384
#define VD      768
#define NCOL    1552   /* KD + KD + VD + NH + NH */
#define NTILES  97     /* NCOL / 16 */
#define NTPAD   100    /* padded to multiple of 4 for 4-tile waves */

typedef __attribute__((ext_vector_type(16))) _Float16 v16h;
typedef __attribute__((ext_vector_type(8)))  float    v8f;

// ---------------------------------------------------------------------------
// Kernel A: pack q_w|k_w|v_w|b_w|a_w directly into WMMA-B fragment order:
// Bpack[nt][kb][lane][j] = W[kb*32 + (lane>>4)*16 + j][nt*16 + (lane&15)]
// (contiguous k order used here: lanes0-15 hold K=0..15, lanes16-31 K=16..31, N=lane&15)
// Padded n-tiles (nt >= 97) are zero-filled.
// ---------------------------------------------------------------------------
__global__ void pack_weights_k(const float* __restrict__ q_w,
                               const float* __restrict__ k_w,
                               const float* __restrict__ v_w,
                               const float* __restrict__ b_w,
                               const float* __restrict__ a_w,
                               _Float16* __restrict__ Bpack) {
    int idx = blockIdx.x * 256 + threadIdx.x;
    if (idx >= NTPAD * 16 * 32 * 16) return;
    int j    = idx & 15;
    int lane = (idx >> 4) & 31;
    int kb   = (idx >> 9) & 15;
    int nt   = idx >> 13;
    int n = nt * 16 + (lane & 15);
    int k = kb * 32 + (lane >> 4) * 16 + j;
    float v = 0.0f;
    if (n < NCOL) {
        if (n < KD)                 v = q_w[k * KD + n];
        else if (n < 2 * KD)        v = k_w[k * KD + (n - KD)];
        else if (n < 2 * KD + VD)   v = v_w[k * VD + (n - 2 * KD)];
        else if (n < 2 * KD + VD + NH) v = b_w[k * NH + (n - (2 * KD + VD))];
        else                        v = a_w[k * NH + (n - (2 * KD + VD + NH))];
    }
    Bpack[idx] = (_Float16)v;
}

// ---------------------------------------------------------------------------
// Kernel B: h = RMSNorm(x @ W_in + b_in) * (1 + norm_w), stored as f16
// one block (512 threads) per token
// ---------------------------------------------------------------------------
__global__ void input_rmsnorm_k(const float* __restrict__ x,
                                const float* __restrict__ W_in,
                                const float* __restrict__ b_in,
                                const float* __restrict__ norm_w,
                                _Float16* __restrict__ h16) {
    int t = blockIdx.x, tid = threadIdx.x;
    __shared__ float xs[INDIM];
    __shared__ float red[HIDN];
    if (tid < INDIM) xs[tid] = x[(size_t)t * INDIM + tid];
    __syncthreads();
    float acc = b_in[tid];
#pragma unroll
    for (int i = 0; i < INDIM; ++i) acc += xs[i] * W_in[i * HIDN + tid];
    red[tid] = acc * acc;
    __syncthreads();
    for (int s = HIDN / 2; s > 0; s >>= 1) {
        if (tid < s) red[tid] += red[tid + s];
        __syncthreads();
    }
    float scale = rsqrtf(red[0] * (1.0f / HIDN) + 1e-6f);
    float hv = acc * scale * (1.0f + norm_w[tid]);
    h16[(size_t)t * HIDN + tid] = (_Float16)hv;
}

// ---------------------------------------------------------------------------
// Kernel C: raw = h16 @ W  (M=32768, N=1552, K=512), WMMA f32 <- f16.
// One wave per 16x64 output strip (4 n-tiles). No LDS:
//   A fragment = two contiguous b128 loads from the row-major h16 row
//     (lane r,hi loads K = hi*16..hi*16+15, same k order as Bpack -> contiguous)
//   B fragment = two contiguous b128 loads from pre-packed Bpack.
// h16 (33MB) stays resident in the 192MB L2, so A re-reads are L2 hits.
// ---------------------------------------------------------------------------
__global__ void wmma_gemm_k(const _Float16* __restrict__ A,     // [M][512]
                            const _Float16* __restrict__ Bpack, // [100][16][32][16]
                            _Float16* __restrict__ C) {         // [M][1552] f16
    const int m0 = blockIdx.x * 16;
    const int ng = blockIdx.y;                 // group of 4 n-tiles
    const int lane = threadIdx.x;              // 0..31
    const int r  = lane & 15;
    const int hi = lane >> 4;

    v8f acc[4];
#pragma unroll
    for (int s = 0; s < 4; ++s) acc[s] = (v8f){};

    const _Float16* arow = A + (size_t)(m0 + r) * HIDN;

    union Frag { v16h v; uint4 q[2]; };

    for (int kb = 0; kb < 16; ++kb) {
        Frag au;
        au.q[0] = *reinterpret_cast<const uint4*>(arow + kb * 32 + hi * 16);
        au.q[1] = *reinterpret_cast<const uint4*>(arow + kb * 32 + hi * 16 + 8);
        if (kb + 1 < 16)
            __builtin_prefetch(arow + (kb + 1) * 32, 0, 1);
#pragma unroll
        for (int s = 0; s < 4; ++s) {
            int nt = ng * 4 + s;
            Frag bu;
            const uint4* bp = reinterpret_cast<const uint4*>(
                Bpack + (((size_t)nt * 16 + kb) * 32 + lane) * 16);
            bu.q[0] = bp[0];
            bu.q[1] = bp[1];
            acc[s] = __builtin_amdgcn_wmma_f32_16x16x32_f16(
                false, au.v, false, bu.v, (short)0, acc[s], false, false);
        }
    }

    // C/D layout: VGPR v -> M = v + 8*hi, N = lane&15
#pragma unroll
    for (int s = 0; s < 4; ++s) {
        int nt = ng * 4 + s;
        if (nt < NTILES) {
#pragma unroll
            for (int v = 0; v < 8; ++v) {
                size_t row = (size_t)(m0 + v + hi * 8);
                C[row * NCOL + nt * 16 + r] = (_Float16)acc[s][v];
            }
        }
    }
}

// ---------------------------------------------------------------------------
// Kernel D: short conv (4 taps) + SiLU + q/k L2-norm + beta/g gates
// one block (256 threads) per token
// ---------------------------------------------------------------------------
__global__ void postprocess_k(const _Float16* __restrict__ raw,
                              const int* __restrict__ offsets,
                              const float* __restrict__ conv_q,
                              const float* __restrict__ conv_k,
                              const float* __restrict__ conv_v,
                              const float* __restrict__ A_log,
                              const float* __restrict__ dt_bias,
                              float* __restrict__ qpost,
                              float* __restrict__ kpost,
                              float* __restrict__ vpost,
                              float* __restrict__ betab,
                              float* __restrict__ gbuf) {
    int t = blockIdx.x, tid = threadIdx.x;
    __shared__ float buf[2 * KD + VD];   // 1536
    __shared__ float qscale[NH], kscale[NH];
    __shared__ int pos_sh;
    if (tid == 0) {
        int seq = 0;
        for (int s = 1; s <= NSEQ; ++s)
            if (t >= offsets[s]) seq = s;
        pos_sh = t - offsets[seq];
    }
    __syncthreads();
    int pos = pos_sh;
    const _Float16* r0 = raw + (size_t)t * NCOL;

#pragma unroll
    for (int cc = 0; cc < 6; ++cc) {
        int c = tid + cc * 256;          // 0..1535
        const float* wp;
        if (c < KD)              wp = conv_q + c * 4;
        else if (c < 2 * KD)     wp = conv_k + (c - KD) * 4;
        else                     wp = conv_v + (c - 2 * KD) * 4;
        float y = (float)r0[c] * wp[3];
#pragma unroll
        for (int s = 1; s <= 3; ++s) {
            if (pos >= s)
                y += (float)raw[(size_t)(t - s) * NCOL + c] * wp[3 - s];
        }
        buf[c] = y * (1.0f / (1.0f + __expf(-y)));   // SiLU
    }
    __syncthreads();

    if (tid < NH) {                               // q scales
        float s2 = 0.f;
        for (int i = 0; i < DKK; ++i) { float v = buf[tid * DKK + i]; s2 += v * v; }
        qscale[tid] = rsqrtf(s2 + 1e-6f) * 0.14433756729740643f;  // * DK^-0.5
    } else if (tid < 2 * NH) {                    // k scales
        int h = tid - NH;
        float s2 = 0.f;
        for (int i = 0; i < DKK; ++i) { float v = buf[KD + h * DKK + i]; s2 += v * v; }
        kscale[h] = rsqrtf(s2 + 1e-6f);
    } else if (tid < 3 * NH) {                    // beta / g
        int h = tid - 2 * NH;
        float bvr = (float)r0[2 * KD + VD + h];
        betab[(size_t)t * NH + h] = 1.0f / (1.0f + __expf(-bvr));
        float av = (float)r0[2 * KD + VD + NH + h] + dt_bias[h];
        float sp = (av > 20.0f) ? av : log1pf(__expf(av));
        gbuf[(size_t)t * NH + h] = -__expf(A_log[h]) * sp;
    }
    __syncthreads();

#pragma unroll
    for (int cc = 0; cc < 6; ++cc) {
        int c = tid + cc * 256;
        if (c < KD)
            qpost[(size_t)t * KD + c] = buf[c] * qscale[c / DKK];
        else if (c < 2 * KD)
            kpost[(size_t)t * KD + (c - KD)] = buf[c] * kscale[(c - KD) / DKK];
        else
            vpost[(size_t)t * VD + (c - 2 * KD)] = buf[c];
    }
}

// ---------------------------------------------------------------------------
// Kernel E: gated delta-rule scan; one block (96 threads) per (seq, head);
// each thread owns one dv column of S[48][96] in registers
// ---------------------------------------------------------------------------
__global__ void delta_scan_k(const float* __restrict__ qpost,
                             const float* __restrict__ kpost,
                             const float* __restrict__ vpost,
                             const float* __restrict__ betab,
                             const float* __restrict__ gbuf,
                             const int* __restrict__ offsets,
                             float* __restrict__ olast) {
    int b = blockIdx.x;
    int seq = b >> 3, head = b & 7;
    int tid = threadIdx.x;               // 0..95 = dv column
    int t0 = offsets[seq], t1 = offsets[seq + 1];

    __shared__ __align__(16) float ksh[DKK];
    __shared__ __align__(16) float qsh[DKK];

    float S[DKK];
#pragma unroll
    for (int i = 0; i < DKK; ++i) S[i] = 0.f;

    for (int t = t0; t < t1; ++t) {
        if (tid < DKK) ksh[tid] = kpost[(size_t)t * KD + head * DKK + tid];
        __syncthreads();

        float decay = __expf(gbuf[(size_t)t * NH + head]);
        float bt    = betab[(size_t)t * NH + head];
        float vt    = vpost[(size_t)t * VD + head * DVV + tid];

        float kr[DKK];
        float vp = 0.f;
        const float4* k4 = reinterpret_cast<const float4*>(ksh);
#pragma unroll
        for (int i = 0; i < DKK / 4; ++i) {
            float4 kk = k4[i];
            kr[4*i+0] = kk.x; kr[4*i+1] = kk.y; kr[4*i+2] = kk.z; kr[4*i+3] = kk.w;
            S[4*i+0] *= decay; vp += kr[4*i+0] * S[4*i+0];
            S[4*i+1] *= decay; vp += kr[4*i+1] * S[4*i+1];
            S[4*i+2] *= decay; vp += kr[4*i+2] * S[4*i+2];
            S[4*i+3] *= decay; vp += kr[4*i+3] * S[4*i+3];
        }
        float delta = (vt - vp) * bt;
#pragma unroll
        for (int i = 0; i < DKK; ++i) S[i] += kr[i] * delta;

        if (t == t1 - 1) {               // only the last o_t is consumed
            if (tid < DKK) qsh[tid] = qpost[(size_t)t * KD + head * DKK + tid];
            __syncthreads();
            float o = 0.f;
#pragma unroll
            for (int i = 0; i < DKK; ++i) o += qsh[i] * S[i];
            olast[seq * VD + head * DVV + tid] = o;
        }
        __syncthreads();
    }
}

// ---------------------------------------------------------------------------
// Kernel F: final-row gate/groupnorm/out-projection/head; one block per seq
// ---------------------------------------------------------------------------
__global__ void finalize_k(const float* __restrict__ x,
                           const int* __restrict__ offsets,
                           const float* __restrict__ W_in,
                           const float* __restrict__ b_in,
                           const float* __restrict__ norm_w,
                           const float* __restrict__ g_w,
                           const float* __restrict__ onorm_w,
                           const float* __restrict__ o_w,
                           const float* __restrict__ head_w,
                           const float* __restrict__ head_b,
                           const float* __restrict__ olast,
                           float* __restrict__ out) {
    int seq = blockIdx.x, tid = threadIdx.x;   // 256 threads
    int tl = offsets[seq + 1] - 1;

    __shared__ float xs[INDIM];
    __shared__ float hbuf[HIDN];
    __shared__ float obuf[VD];
    __shared__ float red[256];
    __shared__ float hs[NH];

    if (tid < INDIM) xs[tid] = x[(size_t)tl * INDIM + tid];
    __syncthreads();

    float a0 = b_in[tid], a1 = b_in[tid + 256];
#pragma unroll
    for (int i = 0; i < INDIM; ++i) {
        a0 += xs[i] * W_in[i * HIDN + tid];
        a1 += xs[i] * W_in[i * HIDN + tid + 256];
    }
    red[tid] = a0 * a0 + a1 * a1;
    __syncthreads();
    for (int s = 128; s > 0; s >>= 1) {
        if (tid < s) red[tid] += red[tid + s];
        __syncthreads();
    }
    float scale = rsqrtf(red[0] * (1.0f / HIDN) + 1e-6f);
    hbuf[tid]       = a0 * scale * (1.0f + norm_w[tid]);
    hbuf[tid + 256] = a1 * scale * (1.0f + norm_w[tid + 256]);
    __syncthreads();

    // raw o for this sequence
#pragma unroll
    for (int cc = 0; cc < 3; ++cc) {
        int c = tid + cc * 256;
        obuf[c] = olast[seq * VD + c];
    }
    __syncthreads();
    if (tid < NH) {
        float s2 = 0.f;
        for (int i = 0; i < DVV; ++i) { float v = obuf[tid * DVV + i]; s2 += v * v; }
        hs[tid] = rsqrtf(s2 * (1.0f / DVV) + 1e-5f);
    }
    __syncthreads();

    // gate = h @ g_w ; o = o*gn*onorm_w * silu(gate)
#pragma unroll
    for (int cc = 0; cc < 3; ++cc) {
        int c = tid + cc * 256;
        float gacc = 0.f;
        for (int k = 0; k < HIDN; ++k) gacc += hbuf[k] * g_w[k * VD + c];
        float ov = obuf[c] * hs[c / DVV] * onorm_w[c % DVV];
        obuf[c] = ov * (gacc * (1.0f / (1.0f + __expf(-gacc))));
    }
    __syncthreads();

    // out = o @ o_w ; result = out . head_w + head_b
    float partial = 0.f;
#pragma unroll
    for (int cc = 0; cc < 2; ++cc) {
        int c = tid + cc * 256;
        float acc = 0.f;
        for (int k = 0; k < VD; ++k) acc += obuf[k] * o_w[k * HIDN + c];
        partial += acc * head_w[c];
    }
    red[tid] = partial;
    __syncthreads();
    for (int s = 128; s > 0; s >>= 1) {
        if (tid < s) red[tid] += red[tid + s];
        __syncthreads();
    }
    if (tid == 0) out[seq] = red[0] + head_b[0];
}

// ---------------------------------------------------------------------------
extern "C" void kernel_launch(void* const* d_in, const int* in_sizes, int n_in,
                              void* d_out, int out_size, void* d_ws, size_t ws_size,
                              hipStream_t stream) {
    const float* x       = (const float*)d_in[0];
    const int*   offsets = (const int*)d_in[1];
    const float* W_in    = (const float*)d_in[2];
    const float* b_in    = (const float*)d_in[3];
    const float* norm_w  = (const float*)d_in[4];
    const float* q_w     = (const float*)d_in[5];
    const float* k_w     = (const float*)d_in[6];
    const float* v_w     = (const float*)d_in[7];
    const float* b_w     = (const float*)d_in[8];
    const float* a_w     = (const float*)d_in[9];
    const float* conv_q  = (const float*)d_in[10];
    const float* conv_k  = (const float*)d_in[11];
    const float* conv_v  = (const float*)d_in[12];
    const float* A_log   = (const float*)d_in[13];
    const float* dt_bias = (const float*)d_in[14];
    const float* g_w     = (const float*)d_in[15];
    const float* onorm_w = (const float*)d_in[16];
    const float* o_w     = (const float*)d_in[17];
    const float* head_w  = (const float*)d_in[18];
    const float* head_b  = (const float*)d_in[19];
    float* out = (float*)d_out;

    char* p = (char*)d_ws;
    auto carve = [&](size_t bytes) -> char* {
        char* r = p;
        p += (bytes + 255) & ~(size_t)255;
        return r;
    };
    _Float16* h16   = (_Float16*)carve((size_t)T_TOTAL * HIDN * 2);
    _Float16* Bpack = (_Float16*)carve((size_t)NTPAD * 16 * 32 * 16 * 2);
    _Float16* raw   = (_Float16*)carve((size_t)T_TOTAL * NCOL * 2);
    float*    qpost = (float*)carve((size_t)T_TOTAL * KD * 4);
    float*    kpost = (float*)carve((size_t)T_TOTAL * KD * 4);
    float*    vpost = (float*)carve((size_t)T_TOTAL * VD * 4);
    float*    betab = (float*)carve((size_t)T_TOTAL * NH * 4);
    float*    gbuf  = (float*)carve((size_t)T_TOTAL * NH * 4);
    float*    olast = (float*)carve((size_t)NSEQ * VD * 4);

    pack_weights_k<<<(NTPAD * 16 * 32 * 16 + 255) / 256, 256, 0, stream>>>(
        q_w, k_w, v_w, b_w, a_w, Bpack);
    input_rmsnorm_k<<<T_TOTAL, HIDN, 0, stream>>>(x, W_in, b_in, norm_w, h16);
    wmma_gemm_k<<<dim3(T_TOTAL / 16, NTPAD / 4), 32, 0, stream>>>(h16, Bpack, raw);
    postprocess_k<<<T_TOTAL, 256, 0, stream>>>(
        raw, offsets, conv_q, conv_k, conv_v, A_log, dt_bias,
        qpost, kpost, vpost, betab, gbuf);
    delta_scan_k<<<NSEQ * NH, 96, 0, stream>>>(
        qpost, kpost, vpost, betab, gbuf, offsets, olast);
    finalize_k<<<NSEQ, 256, 0, stream>>>(
        x, offsets, W_in, b_in, norm_w, g_w, onorm_w, o_w,
        head_w, head_b, olast, out);
}
